// net_13958643712330
// MI455X (gfx1250) — matrix operand order repaired and checked by reading, compiled, not run on hardware
//
#include <hip/hip_runtime.h>

typedef __attribute__((ext_vector_type(16))) __bf16 v16bf;
typedef __attribute__((ext_vector_type(8)))  __bf16 v8bf;
typedef __attribute__((ext_vector_type(8)))  float  v8f;
typedef __attribute__((ext_vector_type(2)))  float  v2f;

#define T_LEN 512
#define B_SZ  4096
#define D_IN  4
#define HID   32

// gfx1250 V_TANH_F32 (confirmed in emitted asm).
#if __has_builtin(__builtin_amdgcn_tanhf)
__device__ __forceinline__ float tanh_f(float x) { return __builtin_amdgcn_tanhf(x); }
#else
__device__ __forceinline__ float tanh_f(float x) {
    float e = __expf(2.0f * x);
    return (e - 1.0f) * __builtin_amdgcn_rcpf(e + 1.0f);
}
#endif
// sigmoid(v + bias) with hb = 0.5*bias pre-scaled: two FMAs + one trans op.
__device__ __forceinline__ float sigm_b(float v, float hb) {
    return fmaf(tanh_f(fmaf(v, 0.5f, hb)), 0.5f, 0.5f);
}

__device__ __forceinline__ v8f wmma_bf16(v16bf a, v16bf b, v8f c) {
    return __builtin_amdgcn_wmma_f32_16x16x32_bf16(false, a, false, b, (short)0, c, false, false);
}
__device__ __forceinline__ v8f wmma_f32k4(v2f a, v2f b, v8f c) {
    return __builtin_amdgcn_wmma_f32_16x16x4_f32(false, a, false, b, (short)0, c, false, false);
}

// Load 16x32 bf16 matrix from row-major LDS into the A operand in this kernel's k order (B uses the same):
// lane row = n; elems 0..15 -> K = 16*hi+0..15.
__device__ __forceinline__ v16bf load_A(const __bf16* base, int n, int hi) {
    const __bf16* p = base + n * HID + 16 * hi;
    v8bf lo = *(const v8bf*)p;
    v8bf hh = *(const v8bf*)(p + 8);
    v16bf a;
    #pragma unroll
    for (int j = 0; j < 8; ++j) { a[j] = lo[j]; a[8 + j] = hh[j]; }
    return a;
}

// 128 threads = 4 waves per block, batch tile of 16 rows:
//   wave 0/1 : layer 1, gate-column half b = 0/1 (producer)
//   wave 2/3 : layer 2, gate-column half b = 0/1 (consumer, one step behind)
// h1 hand-off through a double-buffered LDS tile; ONE barrier per step orders
// everything (b-halves write disjoint columns). grid = B/16 -> 1024 light waves.
__global__ __launch_bounds__(128, 1) void lstm2_pipe4(
    const float* __restrict__ x,
    const float* __restrict__ W_ih0, const float* __restrict__ W_hh0,
    const float* __restrict__ b_ih0, const float* __restrict__ b_hh0,
    const float* __restrict__ W_ih1, const float* __restrict__ W_hh1,
    const float* __restrict__ b_ih1, const float* __restrict__ b_hh1,
    const float* __restrict__ W_out, const float* __restrict__ b_out,
    float* __restrict__ out)
{
    __shared__ __align__(16) __bf16 bufh1[2][16 * HID];   // h1 double buffer
    __shared__ __align__(16) __bf16 bufh2[16 * HID];      // h2 recurrent buffer
    __shared__ __align__(16) float  lds_hf[16 * HID];     // final h2 (f32) for head

    const int tid  = threadIdx.x;
    const int wid  = tid >> 5;        // 0..3
    const int lay  = wid >> 1;        // 0 = layer1, 1 = layer2
    const int bh   = wid & 1;         // gate-column half this wave owns
    const int lane = tid & 31;
    const int n    = lane & 15;       // N column / A row
    const int hi   = lane >> 4;       // K-half selector
    const int row0 = blockIdx.x * 16; // batch tile base
    const v8f zero = {0.f,0.f,0.f,0.f,0.f,0.f,0.f,0.f};

    {   // zero the bf16 h buffers (512 + 256 dwords) with all 128 threads
        unsigned* z1 = (unsigned*)&bufh1[0][0];
        #pragma unroll
        for (int k = 0; k < 4; ++k) z1[tid + 128 * k] = 0u;
        unsigned* z2 = (unsigned*)&bufh2[0];
        #pragma unroll
        for (int k = 0; k < 2; ++k) z2[tid + 128 * k] = 0u;
    }
    __syncthreads();

    if (lay == 0) {
        // ================= layer 1 (this wave: gate tiles class*2 + bh) =================
        v2f   Bih[4];     // f32 4x16 B tiles: V0/V1 = K = 2*hi, 2*hi+1
        v16bf Bhh[4];     // bf16 32x16 B tiles: halves j -> K = 16*hi + j
        float hb[4];
        #pragma unroll
        for (int cls = 0; cls < 4; ++cls) {        // 0=i 1=f 2=g 3=o
            const int gn = (cls * 2 + bh) * 16 + n;
            Bih[cls][0] = W_ih0[gn * D_IN + 2 * hi + 0];
            Bih[cls][1] = W_ih0[gn * D_IN + 2 * hi + 1];
            #pragma unroll
            for (int j = 0; j < 16; ++j)
                Bhh[cls][j] = (__bf16)W_hh0[gn * HID + 16 * hi + j];
            const float bb = b_ih0[gn] + b_hh0[gn];
            hb[cls] = (cls == 2) ? bb : 0.5f * bb; // g-gate keeps full bias
        }
        v8f c1;
        #pragma unroll
        for (int r = 0; r < 8; ++r) c1[r] = 0.f;

        // f32 A-layout for 16x16x4: lane row = n, V0/V1 = K = 2*hi, 2*hi+1
        const float* xrow = x + (size_t)(row0 + n) * T_LEN * D_IN + 2 * hi;
        v2f xv = *(const v2f*)xrow;

        for (int i = 0; i <= T_LEN; ++i) {
            if (i < T_LEN) {
                v16bf Ah1 = load_A(&bufh1[(i + 1) & 1][0], n, hi);  // h1[i-1]
                v2f ax = xv;
                if (i + 1 < T_LEN) xv = *(const v2f*)(xrow + (size_t)(i + 1) * D_IN);
                v8f ai = wmma_f32k4(ax, Bih[0], zero);
                v8f af = wmma_f32k4(ax, Bih[1], zero);
                v8f ag = wmma_f32k4(ax, Bih[2], zero);
                v8f ao = wmma_f32k4(ax, Bih[3], zero);
                ai = wmma_bf16(Ah1, Bhh[0], ai);
                af = wmma_bf16(Ah1, Bhh[1], af);
                ag = wmma_bf16(Ah1, Bhh[2], ag);
                ao = wmma_bf16(Ah1, Bhh[3], ao);
                __bf16* hout = &bufh1[i & 1][0];
                #pragma unroll
                for (int r = 0; r < 8; ++r) {
                    float ig = sigm_b(ai[r], hb[0]);
                    float fg = sigm_b(af[r], hb[1]);
                    float gg = tanh_f(ag[r] + hb[2]);
                    float og = sigm_b(ao[r], hb[3]);
                    float c  = fmaf(fg, c1[r], ig * gg);
                    c1[r] = c;
                    float h  = og * tanh_f(c);
                    hout[(r + 8 * hi) * HID + (n + 16 * bh)] = (__bf16)h;
                }
            }
            __syncthreads();
        }
    } else {
        // ================= layer 2 (one step behind) =================
        v16bf Bih[4], Bhh[4];
        float hb[4];
        #pragma unroll
        for (int cls = 0; cls < 4; ++cls) {
            const int gn = (cls * 2 + bh) * 16 + n;
            #pragma unroll
            for (int j = 0; j < 16; ++j) {
                Bih[cls][j] = (__bf16)W_ih1[gn * HID + 16 * hi + j];
                Bhh[cls][j] = (__bf16)W_hh1[gn * HID + 16 * hi + j];
            }
            const float bb = b_ih1[gn] + b_hh1[gn];
            hb[cls] = (cls == 2) ? bb : 0.5f * bb;
        }
        v8f c2;
        #pragma unroll
        for (int r = 0; r < 8; ++r) c2[r] = 0.f;

        for (int i = 0; i <= T_LEN; ++i) {
            if (i >= 1) {
                v16bf Ah1 = load_A(&bufh1[(i + 1) & 1][0], n, hi);  // h1[i-1]
                v16bf Ah2 = load_A(&bufh2[0], n, hi);               // h2[i-2]
                v8f ai = wmma_bf16(Ah1, Bih[0], zero);
                v8f af = wmma_bf16(Ah1, Bih[1], zero);
                v8f ag = wmma_bf16(Ah1, Bih[2], zero);
                v8f ao = wmma_bf16(Ah1, Bih[3], zero);
                ai = wmma_bf16(Ah2, Bhh[0], ai);
                af = wmma_bf16(Ah2, Bhh[1], af);
                ag = wmma_bf16(Ah2, Bhh[2], ag);
                ao = wmma_bf16(Ah2, Bhh[3], ao);
                #pragma unroll
                for (int r = 0; r < 8; ++r) {
                    float ig = sigm_b(ai[r], hb[0]);
                    float fg = sigm_b(af[r], hb[1]);
                    float gg = tanh_f(ag[r] + hb[2]);
                    float og = sigm_b(ao[r], hb[3]);
                    float c  = fmaf(fg, c2[r], ig * gg);
                    c2[r] = c;
                    float h  = og * tanh_f(c);
                    bufh2[(r + 8 * hi) * HID + (n + 16 * bh)] = (__bf16)h;
                    if (i == T_LEN)
                        lds_hf[(r + 8 * hi) * HID + (n + 16 * bh)] = h;
                }
            }
            __syncthreads();
        }
        // ---- output head: out[b] = h2_last . W_out + b_out (f32) ----
        // lds_hf fully written (both b-halves) before the final loop barrier.
        if (bh == 0 && hi == 0) {
            float s = b_out[0];
            #pragma unroll
            for (int k = 0; k < HID; ++k) s += lds_hf[n * HID + k] * W_out[k];
            out[row0 + n] = s;
        }
    }
}

extern "C" void kernel_launch(void* const* d_in, const int* in_sizes, int n_in,
                              void* d_out, int out_size, void* d_ws, size_t ws_size,
                              hipStream_t stream) {
    const float* x     = (const float*)d_in[0];
    const float* W_ih0 = (const float*)d_in[1];
    const float* W_hh0 = (const float*)d_in[2];
    const float* b_ih0 = (const float*)d_in[3];
    const float* b_hh0 = (const float*)d_in[4];
    const float* W_ih1 = (const float*)d_in[5];
    const float* W_hh1 = (const float*)d_in[6];
    const float* b_ih1 = (const float*)d_in[7];
    const float* b_hh1 = (const float*)d_in[8];
    const float* W_out = (const float*)d_in[9];
    const float* b_out = (const float*)d_in[10];
    float* out = (float*)d_out;

    dim3 grid(B_SZ / 16);   // 256 batch tiles
    dim3 block(128);        // 4 light waves: (layer, gate-column half)
    hipLaunchKernelGGL(lstm2_pipe4, grid, block, 0, stream,
                       x, W_ih0, W_hh0, b_ih0, b_hh0,
                       W_ih1, W_hh1, b_ih1, b_hh1, W_out, b_out, out);
}
